// SparseGraphAttentionNetwork_1288490189384
// MI455X (gfx1250) — matrix operand order repaired and checked by reading, not compiled, not run on hardware
//
#include <hip/hip_runtime.h>

#define N_NODES 100000
#define N_EDGES 1600000
#define IN_F    128
#define HID     64
#define OUTF    64
#define NHEADS  8
#define HCAT    512   // NHEADS*HID

typedef __attribute__((ext_vector_type(16))) _Float16 v16h;
typedef __attribute__((ext_vector_type(8)))  _Float16 v8h;
typedef __attribute__((ext_vector_type(2)))  _Float16 v2h;
typedef __attribute__((ext_vector_type(8)))  float    v8f;

// ---- WMMA fragment loaders (16x16x32 f16, wave32, contiguous k order shared by A and B) ----
// A (16x32, row-major in memory): lane<16 holds row m, K=0..15; lane>=16 holds K=16..31.
__device__ __forceinline__ v16h load_a_frag(const _Float16* __restrict__ A, int rowlen,
                                            int m, int k0, int half) {
  return *(const v16h*)(A + (size_t)m * rowlen + k0 + half * 16);
}
// B (32x16): stored TRANSPOSED (Bt[n][k]) so lane (n=lane&15, half=lane>>4) reads 16 contiguous halves.
__device__ __forceinline__ v16h load_b_frag(const _Float16* __restrict__ Bt, int rowlen,
                                            int n, int k0, int half) {
  return *(const v16h*)(Bt + (size_t)n * rowlen + k0 + half * 16);
}

// ---- GEMM 1: xp[N,128] = f16(x)[N,128] @ W0t + b0, output f16 ----
__global__ void gemm_xp(const _Float16* __restrict__ Ah, const _Float16* __restrict__ W0t,
                        const float* __restrict__ b0, _Float16* __restrict__ xph) {
  int wave = threadIdx.x >> 5, lane = threadIdx.x & 31;
  int half = lane >> 4, lm = lane & 15;
  int mBase = blockIdx.x * 16;
  int n = wave * 16 + lm;
  int m = mBase + lm;
  v8f c = {};
#pragma unroll
  for (int kt = 0; kt < 4; kt++) {
    v16h a = load_a_frag(Ah, IN_F, m, kt * 32, half);
    v16h b = load_b_frag(W0t, IN_F, n, kt * 32, half);
    c = __builtin_amdgcn_wmma_f32_16x16x32_f16(false, a, false, b, (short)0, c, false, false);
  }
  float bn = b0[n];
#pragma unroll
  for (int v = 0; v < 8; v++) {
    int row = mBase + half * 8 + v;
    xph[(size_t)row * IN_F + n] = (_Float16)(c[v] + bn);
  }
}

// ---- GEMM 2: h_all[N,512] f16 = xp_f16[N,128] @ WcatT (all 8 heads fused) ----
// f16 output keeps the edge-gather table at 102 MB (< 192 MB L2).
__global__ void gemm_heads(const _Float16* __restrict__ Ah, const _Float16* __restrict__ WcatT,
                           _Float16* __restrict__ h_all) {
  int wave = threadIdx.x >> 5, lane = threadIdx.x & 31;
  int half = lane >> 4, lm = lane & 15;
  int mBase = blockIdx.x * 16;
  int m = mBase + lm;
  v16h afrag[4];
#pragma unroll
  for (int kt = 0; kt < 4; kt++) afrag[kt] = load_a_frag(Ah, IN_F, m, kt * 32, half);
#pragma unroll
  for (int t = 0; t < 4; t++) {
    int n = (wave * 4 + t) * 16 + lm;
    v8f c = {};
#pragma unroll
    for (int kt = 0; kt < 4; kt++) {
      v16h b = load_b_frag(WcatT, IN_F, n, kt * 32, half);
      c = __builtin_amdgcn_wmma_f32_16x16x32_f16(false, afrag[kt], false, b, (short)0, c, false, false);
    }
#pragma unroll
    for (int v = 0; v < 8; v++)
      h_all[(size_t)(mBase + half * 8 + v) * HCAT + n] = (_Float16)c[v];
  }
}

// ---- GEMM 3: h2[N,64] f16 = h1_f16[N,512] @ WendT ----
__global__ void gemm_end(const _Float16* __restrict__ Ah, const _Float16* __restrict__ WendT,
                         _Float16* __restrict__ h2) {
  int wave = threadIdx.x >> 5, lane = threadIdx.x & 31;
  int half = lane >> 4, lm = lane & 15;
  int mBase = blockIdx.x * 32 + (wave >> 2) * 16;
  int n = (wave & 3) * 16 + lm;
  int m = mBase + lm;
  v8f c = {};
#pragma unroll 4
  for (int kt = 0; kt < 16; kt++) {
    v16h a = load_a_frag(Ah, HCAT, m, kt * 32, half);
    v16h b = load_b_frag(WendT, HCAT, n, kt * 32, half);
    c = __builtin_amdgcn_wmma_f32_16x16x32_f16(false, a, false, b, (short)0, c, false, false);
  }
#pragma unroll
  for (int v = 0; v < 8; v++)
    h2[(size_t)(mBase + half * 8 + v) * OUTF + n] = (_Float16)c[v];
}

// ---- helpers ----
__global__ void zero_f4(float4* __restrict__ p, long n4) {
  long i = (long)blockIdx.x * blockDim.x + threadIdx.x;
  long stride = (long)gridDim.x * blockDim.x;
  for (; i < n4; i += stride) p[i] = make_float4(0.f, 0.f, 0.f, 0.f);
}
__global__ void f32_to_f16(const float* __restrict__ in, _Float16* __restrict__ out, long n) {
  long i = (long)blockIdx.x * blockDim.x + threadIdx.x;
  if (i < n) out[i] = (_Float16)in[i];
}
// in[rows][cols] -> out[cols][rows] as f16
__global__ void transpose_to_f16(const float* __restrict__ in, _Float16* __restrict__ out,
                                 int rows, int cols) {
  int i = blockIdx.x * blockDim.x + threadIdx.x;
  if (i >= rows * cols) return;
  int o = i / rows, r = i - o * rows;
  out[(size_t)o * rows + r] = (_Float16)in[(size_t)r * cols + o];
}
// W_heads[8][128][64] -> WcatT[512][128] f16 (column-major concat of heads)
__global__ void prep_wcat(const float* __restrict__ Wh, _Float16* __restrict__ out) {
  int i = blockIdx.x * blockDim.x + threadIdx.x;
  if (i >= HCAT * IN_F) return;
  int o = i >> 7, r = i & 127;
  int head = o >> 6, fo = o & 63;
  out[i] = (_Float16)Wh[((size_t)head * IN_F + r) * HID + fo];
}

// ---- per-node attention scores (read f16 feature tables, f32 math) ----
__global__ void scores1(const _Float16* __restrict__ h_all, const float* __restrict__ a_heads,
                        float* __restrict__ ssrc, float* __restrict__ sdst) {
  int idx = blockIdx.x * blockDim.x + threadIdx.x;
  if (idx >= N_NODES * NHEADS) return;
  int n = idx >> 3, h = idx & 7;
  const _Float16* hp = h_all + (size_t)n * HCAT + h * HID;
  const float* av = a_heads + h * 2 * HID;
  float ss = 0.f, sd = 0.f;
#pragma unroll 8
  for (int f = 0; f < HID; f++) {
    float xv = (float)hp[f];
    ss += xv * av[f]; sd += xv * av[HID + f];
  }
  ssrc[idx] = ss; sdst[idx] = sd;
}
__global__ void scores2(const _Float16* __restrict__ h2, const float* __restrict__ a_end,
                        float* __restrict__ ssrc, float* __restrict__ sdst) {
  int n = blockIdx.x * blockDim.x + threadIdx.x;
  if (n >= N_NODES) return;
  const _Float16* hp = h2 + (size_t)n * OUTF;
  float ss = 0.f, sd = 0.f;
#pragma unroll 8
  for (int f = 0; f < OUTF; f++) {
    float xv = (float)hp[f];
    ss += xv * a_end[f]; sd += xv * a_end[OUTF + f];
  }
  ssrc[n] = ss; sdst[n] = sd;
}

// ---- edge aggregation, layer 1 (all 8 heads): one wave per edge, 16 f16/lane gather, f32 atomics ----
__global__ void edge_agg1(const int* __restrict__ edges, const _Float16* __restrict__ h_all,
                          const float* __restrict__ ssrc, const float* __restrict__ sdst,
                          float* __restrict__ agg, float* __restrict__ rowsum) {
  int wave = threadIdx.x >> 5, lane = threadIdx.x & 31;
  int e = blockIdx.x * 8 + wave;
  int s = edges[e], d = edges[N_EDGES + e];
  int head = lane >> 2;  // lane covers features [lane*16, lane*16+16) -> single head
  float sc = ssrc[s * NHEADS + head] + sdst[d * NHEADS + head];
  float w = __expf(-(sc > 0.f ? sc : 0.2f * sc));
  const _Float16* hd = h_all + (size_t)d * HCAT + lane * 16;
  v8h v0 = *(const v8h*)(hd);
  v8h v1 = *(const v8h*)(hd + 8);
  float* ag = agg + (size_t)s * HCAT + lane * 16;
#pragma unroll
  for (int j = 0; j < 8; j++) {
    atomicAdd(ag + j,     w * (float)v0[j]);
    atomicAdd(ag + 8 + j, w * (float)v1[j]);
  }
  if (lane < NHEADS) {
    float sc2 = ssrc[s * NHEADS + lane] + sdst[d * NHEADS + lane];
    float w2 = __expf(-(sc2 > 0.f ? sc2 : 0.2f * sc2));
    atomicAdd(rowsum + s * NHEADS + lane, w2);
  }
}

// ---- normalize + elu -> f16 for next GEMM ----
__global__ void finalize1(const float* __restrict__ agg, const float* __restrict__ rowsum,
                          _Float16* __restrict__ h1h) {
  long i4 = (long)blockIdx.x * blockDim.x + threadIdx.x;
  if (i4 >= (long)N_NODES * HCAT / 4) return;
  long f0 = i4 * 4;
  int n = (int)(f0 >> 9);
  int f = (int)(f0 & 511);
  float r = 1.f / (rowsum[n * NHEADS + (f >> 6)] + 1e-16f);
  float4 v = ((const float4*)agg)[i4];
  float o[4] = {v.x * r, v.y * r, v.z * r, v.w * r};
  _Float16* out = h1h + f0;
#pragma unroll
  for (int j = 0; j < 4; j++) {
    float xv = o[j];
    out[j] = (_Float16)(xv > 0.f ? xv : (__expf(xv) - 1.f));
  }
}

// ---- edge aggregation, layer 2 (64 features): one wave per edge, 2 f16/lane gather ----
__global__ void edge_agg2(const int* __restrict__ edges, const _Float16* __restrict__ h2,
                          const float* __restrict__ ssrc, const float* __restrict__ sdst,
                          float* __restrict__ agg2, float* __restrict__ rowsum2) {
  int wave = threadIdx.x >> 5, lane = threadIdx.x & 31;
  int e = blockIdx.x * 8 + wave;
  int s = edges[e], d = edges[N_EDGES + e];
  float sc = ssrc[s] + sdst[d];
  float w = __expf(-(sc > 0.f ? sc : 0.2f * sc));
  v2h v = ((const v2h*)(h2 + (size_t)d * OUTF))[lane];
  float* ag = agg2 + (size_t)s * OUTF + lane * 2;
  atomicAdd(ag + 0, w * (float)v[0]);
  atomicAdd(ag + 1, w * (float)v[1]);
  if (lane == 0) atomicAdd(rowsum2 + s, w);
}

// ---- normalize + row softmax (wave32 shuffle reduction) ----
__global__ void finalize2(const float* __restrict__ agg2, const float* __restrict__ rowsum2,
                          float* __restrict__ out) {
  int wave = threadIdx.x >> 5, lane = threadIdx.x & 31;
  int n = blockIdx.x * 8 + wave;
  if (n >= N_NODES) return;
  float r = 1.f / (rowsum2[n] + 1e-16f);
  float2 v = ((const float2*)(agg2 + (size_t)n * OUTF))[lane];
  float a0 = v.x * r, a1 = v.y * r;
  float m = fmaxf(a0, a1);
#pragma unroll
  for (int off = 16; off > 0; off >>= 1) m = fmaxf(m, __shfl_xor(m, off, 32));
  float e0 = __expf(a0 - m), e1 = __expf(a1 - m);
  float sum = e0 + e1;
#pragma unroll
  for (int off = 16; off > 0; off >>= 1) sum += __shfl_xor(sum, off, 32);
  float inv = 1.f / sum;
  ((float2*)(out + (size_t)n * OUTF))[lane] = make_float2(e0 * inv, e1 * inv);
}

extern "C" void kernel_launch(void* const* d_in, const int* in_sizes, int n_in,
                              void* d_out, int out_size, void* d_ws, size_t ws_size,
                              hipStream_t stream) {
  (void)in_sizes; (void)n_in; (void)out_size; (void)ws_size;
  const float* x       = (const float*)d_in[0];
  const int*   edges   = (const int*)d_in[1];
  const float* W0      = (const float*)d_in[2];
  const float* b0      = (const float*)d_in[3];
  const float* W_heads = (const float*)d_in[4];
  const float* a_heads = (const float*)d_in[5];
  const float* W_end   = (const float*)d_in[6];
  const float* a_end   = (const float*)d_in[7];
  float* out = (float*)d_out;

  char* ws = (char*)d_ws;
  size_t off = 0;
  auto alloc = [&](size_t bytes) -> void* {
    void* p = ws + off;
    off += (bytes + 255) & ~(size_t)255;
    return p;
  };
  // accumulators first (contiguous -> single zero pass)
  float*    agg     = (float*)alloc((size_t)N_NODES * HCAT * 4);
  float*    rowsum  = (float*)alloc((size_t)N_NODES * NHEADS * 4);
  float*    agg2    = (float*)alloc((size_t)N_NODES * OUTF * 4);
  float*    rowsum2 = (float*)alloc((size_t)N_NODES * 4);
  size_t    zero_bytes = off;
  _Float16* xh      = (_Float16*)alloc((size_t)N_NODES * IN_F * 2);
  _Float16* xph     = (_Float16*)alloc((size_t)N_NODES * IN_F * 2);
  _Float16* h_all   = (_Float16*)alloc((size_t)N_NODES * HCAT * 2);
  float*    ssrc    = (float*)alloc((size_t)N_NODES * NHEADS * 4);
  float*    sdst    = (float*)alloc((size_t)N_NODES * NHEADS * 4);
  _Float16* h1h     = (_Float16*)alloc((size_t)N_NODES * HCAT * 2);
  _Float16* h2      = (_Float16*)alloc((size_t)N_NODES * OUTF * 2);
  float*    s2s     = (float*)alloc((size_t)N_NODES * 4);
  float*    s2d     = (float*)alloc((size_t)N_NODES * 4);
  _Float16* w0t     = (_Float16*)alloc((size_t)IN_F * IN_F * 2);
  _Float16* wcatT   = (_Float16*)alloc((size_t)HCAT * IN_F * 2);
  _Float16* wendT   = (_Float16*)alloc((size_t)OUTF * HCAT * 2);

  const int TB = 256;
  zero_f4<<<4096, TB, 0, stream>>>((float4*)agg, (long)(zero_bytes / 16));

  f32_to_f16<<<(N_NODES * IN_F + TB - 1) / TB, TB, 0, stream>>>(x, xh, (long)N_NODES * IN_F);
  transpose_to_f16<<<(IN_F * IN_F + TB - 1) / TB, TB, 0, stream>>>(W0, w0t, IN_F, IN_F);
  prep_wcat<<<(HCAT * IN_F + TB - 1) / TB, TB, 0, stream>>>(W_heads, wcatT);
  transpose_to_f16<<<(HCAT * OUTF + TB - 1) / TB, TB, 0, stream>>>(W_end, wendT, HCAT, OUTF);

  gemm_xp<<<N_NODES / 16, TB, 0, stream>>>(xh, w0t, b0, xph);
  gemm_heads<<<N_NODES / 16, TB, 0, stream>>>(xph, wcatT, h_all);
  scores1<<<(N_NODES * NHEADS + TB - 1) / TB, TB, 0, stream>>>(h_all, a_heads, ssrc, sdst);
  edge_agg1<<<N_EDGES / 8, TB, 0, stream>>>(edges, h_all, ssrc, sdst, agg, rowsum);
  finalize1<<<(int)(((long)N_NODES * HCAT / 4 + TB - 1) / TB), TB, 0, stream>>>(agg, rowsum, h1h);

  gemm_end<<<N_NODES / 32, TB, 0, stream>>>(h1h, wendT, h2);
  scores2<<<(N_NODES + TB - 1) / TB, TB, 0, stream>>>(h2, a_end, s2s, s2d);
  edge_agg2<<<N_EDGES / 8, TB, 0, stream>>>(edges, h2, s2s, s2d, agg2, rowsum2);
  finalize2<<<N_NODES / 8, TB, 0, stream>>>(agg2, rowsum2, out);
}
